// NonLocalSelfSimilarity_24962349924866
// MI455X (gfx1250) — matrix operand order repaired and checked by reading, not compiled, not run on hardware
//
#include <hip/hip_runtime.h>
#include <cstdint>
#include <cstddef>

#define BB 4
#define CC 256
#define CR 32
#define NN 4096
#define EPSV 1e-5f

typedef __attribute__((ext_vector_type(16))) _Float16 v16h;
typedef __attribute__((ext_vector_type(8)))  float    v8f;
typedef __attribute__((ext_vector_type(4)))  unsigned int u32x4;

union V16 {
  v16h  h;
  u32x4 u[2];
};

static __device__ __forceinline__ v8f wmma_f16(const V16& a, const V16& b, v8f c) {
  // D = A(16x32 f16) * B(32x16 f16) + C(16x16 f32)
  return __builtin_amdgcn_wmma_f32_16x16x32_f16(false, a.h, false, b.h,
                                                (short)0, c, false, false);
}

// ---------------------------------------------------------------------------
// Kernel 0a: transpose+convert x [B][C][N] f32 -> xT [B][N][C] f16
// ---------------------------------------------------------------------------
__global__ void k_prep_x(const float* __restrict__ x, _Float16* __restrict__ xT) {
  int blk = blockIdx.x;
  int nt  = blk % (NN / 16);
  int b   = blk / (NN / 16);
  int tx = threadIdx.x & 15, ty = threadIdx.x >> 4;
  __shared__ float tile[16][17];
  for (int c0 = 0; c0 < CC; c0 += 16) {
    tile[ty][tx] = x[((size_t)(b * CC + c0 + ty)) * NN + nt * 16 + tx];
    __syncthreads();
    xT[((size_t)(b * NN + nt * 16 + ty)) * CC + c0 + tx] = (_Float16)tile[tx][ty];
    __syncthreads();
  }
}

// ---------------------------------------------------------------------------
// Kernel 0b: convert weights to f16 row-major; fold BN affine into sA/sB
// ---------------------------------------------------------------------------
__global__ void k_prep_params(const float* __restrict__ Wq, const float* __restrict__ Wk,
                              const float* __restrict__ Wv, const float* __restrict__ Wo,
                              const float* __restrict__ bo, const float* __restrict__ gamma,
                              const float* __restrict__ beta, const float* __restrict__ rmean,
                              const float* __restrict__ rvar,
                              _Float16* __restrict__ wq16, _Float16* __restrict__ wk16,
                              _Float16* __restrict__ wv16, _Float16* __restrict__ wo16,
                              float* __restrict__ sA, float* __restrict__ sB) {
  int i = blockIdx.x * blockDim.x + threadIdx.x;
  if (i < CR * CC) { wq16[i] = (_Float16)Wq[i]; wk16[i] = (_Float16)Wk[i]; }
  if (i < CC * CC) { wv16[i] = (_Float16)Wv[i]; wo16[i] = (_Float16)Wo[i]; }
  if (i < CC) {
    float inv = gamma[i] * rsqrtf(rvar[i] + EPSV);
    sA[i] = inv;
    sB[i] = (bo[i] - rmean[i]) * inv + beta[i];
  }
}

// ---------------------------------------------------------------------------
// Kernel 1: QKV projections. One wave per (b, n-tile, output-tile).
// ot 0-1 -> Q (Qt [B][N][CR]), ot 2-3 -> K (Kt [B][N][CR]), ot 4-19 -> V ([B][C][N])
// ---------------------------------------------------------------------------
__global__ void k_qkv(const _Float16* __restrict__ xT,
                      const _Float16* __restrict__ wq16, const _Float16* __restrict__ wk16,
                      const _Float16* __restrict__ wv16,
                      const float* __restrict__ bq, const float* __restrict__ bk,
                      const float* __restrict__ bv,
                      _Float16* __restrict__ Qt, _Float16* __restrict__ Kt,
                      _Float16* __restrict__ Vh) {
  int wid = blockIdx.x;
  int ot  = wid % 20;
  int nt  = (wid / 20) % (NN / 16);
  int b   = wid / (20 * (NN / 16));
  int l = threadIdx.x, lh = l >> 4, lm = l & 15;

  const _Float16* wrow;
  if (ot < 2)      wrow = wq16 + ((size_t)(ot * 16 + lm)) * CC;
  else if (ot < 4) wrow = wk16 + ((size_t)((ot - 2) * 16 + lm)) * CC;
  else             wrow = wv16 + ((size_t)((ot - 4) * 16 + lm)) * CC;
  const _Float16* xrow = xT + ((size_t)(b * NN + nt * 16 + lm)) * CC;

  v8f acc = {};
#pragma unroll
  for (int kc = 0; kc < CC; kc += 32) {
    V16 a, bm;
    a.u[0]  = *(const u32x4*)(wrow + kc + lh * 8);
    a.u[1]  = *(const u32x4*)(wrow + kc + 16 + lh * 8);
    bm.u[0] = *(const u32x4*)(xrow + kc + lh * 8);
    bm.u[1] = *(const u32x4*)(xrow + kc + 16 + lh * 8);
    acc = wmma_f16(a, bm, acc);
  }

  int n = nt * 16 + lm;  // D: column N = lane%16, row M = r + 8*(lane/16)
  if (ot < 4) {
    const float* bias = (ot < 2) ? bq : bk;
    int coff = (ot & 1) * 16;
    _Float16* dst = (ot < 2 ? Qt : Kt) + ((size_t)(b * NN + n)) * CR + coff;
#pragma unroll
    for (int r = 0; r < 8; ++r) {
      int m = r + 8 * lh;
      dst[m] = (_Float16)(acc[r] + bias[coff + m]);
    }
  } else {
    int cbase = (ot - 4) * 16;
#pragma unroll
    for (int r = 0; r < 8; ++r) {
      int m = r + 8 * lh;
      Vh[((size_t)(b * CC + cbase + m)) * NN + n] = (_Float16)(acc[r] + bv[cbase + m]);
    }
  }
}

// ---------------------------------------------------------------------------
// Kernel 2: fused flash attention. One wave per (b, 16-query tile, 64-channel
// quarter). Never materializes attn; P goes regs -> LDS (transpose) -> B-operand.
// ---------------------------------------------------------------------------
__global__ void k_attn(const _Float16* __restrict__ Qt, const _Float16* __restrict__ Kt,
                       const _Float16* __restrict__ Vh, _Float16* __restrict__ AO) {
  int wid = blockIdx.x;
  int cb  = wid & 3;
  int qt  = (wid >> 2) % (NN / 16);
  int b   = wid / (4 * (NN / 16));
  int l = threadIdx.x, lh = l >> 4, lm = l & 15;

  __shared__ __align__(16) _Float16 Pb[16 * 32];  // [q row][k col]
  __shared__ float den[16];

  // A operand: Q tile 16(q) x 32(c), reused for whole key loop
  V16 aq;
  const _Float16* qrow = Qt + ((size_t)(b * NN + qt * 16 + lm)) * CR;
  aq.u[0] = *(const u32x4*)(qrow + lh * 8);
  aq.u[1] = *(const u32x4*)(qrow + 16 + lh * 8);

  v8f acc[4] = {};
  float dsum[8] = {0.f, 0.f, 0.f, 0.f, 0.f, 0.f, 0.f, 0.f};
  const float kls = 0.0625f * 1.4426950408889634f;  // C^-0.5 * log2(e)

  const _Float16* Kb = Kt + ((size_t)b * NN) * CR;
  const _Float16* Vb = Vh + ((size_t)b * CC) * NN;

  for (int kc = 0; kc < NN; kc += 32) {
    // logits S = Q^T K for 16q x 32k
    V16 bk0, bk1;
    const _Float16* k0 = Kb + (size_t)(kc + lm) * CR;
    const _Float16* k1 = Kb + (size_t)(kc + 16 + lm) * CR;
    bk0.u[0] = ((const u32x4*)k0)[lh]; bk0.u[1] = ((const u32x4*)k0)[2 + lh];
    bk1.u[0] = ((const u32x4*)k1)[lh]; bk1.u[1] = ((const u32x4*)k1)[2 + lh];
    v8f z = {};
    v8f s0 = wmma_f16(aq, bk0, z);
    v8f s1 = wmma_f16(aq, bk1, z);

    // P = exp(S/16); accumulate per-lane denominator partials; transpose via LDS
#pragma unroll
    for (int r = 0; r < 8; ++r) {
      float p0 = exp2f(s0[r] * kls);
      float p1 = exp2f(s1[r] * kls);
      dsum[r] += p0 + p1;
      int m = r + 8 * lh;                 // q row of this element
      Pb[m * 32 + lm]      = (_Float16)p0;  // k = lm
      Pb[m * 32 + lm + 16] = (_Float16)p1;  // k = lm + 16
    }
    __syncthreads();  // single-wave WG: compiles to a DS-count wait

    // B operand for PV: P^T tile, 32(k) x 16(q): lane = column q, k = 8h+s / 16+8h+(s-8)
    V16 bp;
    const _Float16* prow = Pb + lm * 32;
    bp.u[0] = ((const u32x4*)prow)[lh];
    bp.u[1] = ((const u32x4*)prow)[2 + lh];

    // acc[c-tile] += V(16c x 32k) * P^T(32k x 16q)
#pragma unroll
    for (int ct = 0; ct < 4; ++ct) {
      const _Float16* vrow = Vb + ((size_t)(cb * 64 + ct * 16 + lm)) * NN + kc;
      V16 av;
      av.u[0] = *(const u32x4*)(vrow + lh * 8);
      av.u[1] = *(const u32x4*)(vrow + 16 + lh * 8);
      acc[ct] = wmma_f16(av, bp, acc[ct]);
    }
  }

  // full row-sum reduction across the 16 lanes of each half-wave
#pragma unroll
  for (int r = 0; r < 8; ++r) {
    float v = dsum[r];
    v += __shfl_xor(v, 1);
    v += __shfl_xor(v, 2);
    v += __shfl_xor(v, 4);
    v += __shfl_xor(v, 8);
    dsum[r] = v;
  }
  if (lm == 0) {
#pragma unroll
    for (int r = 0; r < 8; ++r) den[r + 8 * lh] = dsum[r];
  }
  __syncthreads();

  float rden = 1.0f / den[lm];  // PV output column N == q == lane%16
  _Float16* aorow = AO + ((size_t)(b * NN + qt * 16 + lm)) * CC + cb * 64;
#pragma unroll
  for (int ct = 0; ct < 4; ++ct) {
#pragma unroll
    for (int r = 0; r < 8; ++r)
      aorow[ct * 16 + r + 8 * lh] = (_Float16)(acc[ct][r] * rden);
  }
}

// ---------------------------------------------------------------------------
// Kernel 3: Wo projection + folded BN + residual + ReLU
// ---------------------------------------------------------------------------
__global__ void k_out(const _Float16* __restrict__ AO, const _Float16* __restrict__ wo16,
                      const float* __restrict__ sA, const float* __restrict__ sB,
                      const float* __restrict__ x, float* __restrict__ out) {
  int wid = blockIdx.x;
  int ot  = wid % (CC / 16);
  int nt  = (wid / (CC / 16)) % (NN / 16);
  int b   = wid / ((CC / 16) * (NN / 16));
  int l = threadIdx.x, lh = l >> 4, lm = l & 15;

  const _Float16* wrow = wo16 + ((size_t)(ot * 16 + lm)) * CC;
  const _Float16* arow = AO + ((size_t)(b * NN + nt * 16 + lm)) * CC;
  v8f acc = {};
#pragma unroll
  for (int kc = 0; kc < CC; kc += 32) {
    V16 a, bm;
    a.u[0]  = *(const u32x4*)(wrow + kc + lh * 8);
    a.u[1]  = *(const u32x4*)(wrow + kc + 16 + lh * 8);
    bm.u[0] = *(const u32x4*)(arow + kc + lh * 8);
    bm.u[1] = *(const u32x4*)(arow + kc + 16 + lh * 8);
    acc = wmma_f16(a, bm, acc);
  }
  int n = nt * 16 + lm;
#pragma unroll
  for (int r = 0; r < 8; ++r) {
    int o = ot * 16 + r + 8 * lh;
    size_t xi = ((size_t)(b * CC + o)) * NN + n;
    float v = acc[r] * sA[o] + sB[o] + x[xi];
    out[xi] = fmaxf(v, 0.0f);
  }
}

// ---------------------------------------------------------------------------
extern "C" void kernel_launch(void* const* d_in, const int* in_sizes, int n_in,
                              void* d_out, int out_size, void* d_ws, size_t ws_size,
                              hipStream_t stream) {
  (void)in_sizes; (void)n_in; (void)out_size; (void)ws_size;
  const float* x     = (const float*)d_in[0];
  const float* Wq    = (const float*)d_in[1];
  const float* bq    = (const float*)d_in[2];
  const float* Wk    = (const float*)d_in[3];
  const float* bk    = (const float*)d_in[4];
  const float* Wv    = (const float*)d_in[5];
  const float* bv    = (const float*)d_in[6];
  const float* Wo    = (const float*)d_in[7];
  const float* bo    = (const float*)d_in[8];
  const float* gamma = (const float*)d_in[9];
  const float* beta  = (const float*)d_in[10];
  const float* rmean = (const float*)d_in[11];
  const float* rvar  = (const float*)d_in[12];
  float* out = (float*)d_out;

  // workspace carve-out (~26.3 MB total), 256B aligned regions
  char* w = (char*)d_ws;
  auto alloc = [&](size_t bytes) {
    void* p = (void*)w;
    w += (bytes + 255) & ~(size_t)255;
    return p;
  };
  _Float16* xT   = (_Float16*)alloc((size_t)BB * NN * CC * 2);
  _Float16* wq16 = (_Float16*)alloc((size_t)CR * CC * 2);
  _Float16* wk16 = (_Float16*)alloc((size_t)CR * CC * 2);
  _Float16* wv16 = (_Float16*)alloc((size_t)CC * CC * 2);
  _Float16* wo16 = (_Float16*)alloc((size_t)CC * CC * 2);
  _Float16* Qt   = (_Float16*)alloc((size_t)BB * NN * CR * 2);
  _Float16* Kt   = (_Float16*)alloc((size_t)BB * NN * CR * 2);
  _Float16* Vh   = (_Float16*)alloc((size_t)BB * CC * NN * 2);
  _Float16* AO   = (_Float16*)alloc((size_t)BB * NN * CC * 2);
  float* sA = (float*)alloc((size_t)CC * 4);
  float* sB = (float*)alloc((size_t)CC * 4);

  k_prep_x<<<BB * (NN / 16), 256, 0, stream>>>(x, xT);
  k_prep_params<<<(CC * CC) / 256, 256, 0, stream>>>(Wq, Wk, Wv, Wo, bo, gamma, beta,
                                                     rmean, rvar, wq16, wk16, wv16, wo16,
                                                     sA, sB);
  k_qkv<<<BB * (NN / 16) * 20, 32, 0, stream>>>(xT, wq16, wk16, wv16, bq, bk, bv, Qt, Kt, Vh);
  k_attn<<<BB * (NN / 16) * 4, 32, 0, stream>>>(Qt, Kt, Vh, AO);
  k_out<<<BB * (NN / 16) * (CC / 16), 32, 0, stream>>>(AO, wo16, sA, sB, x, out);
}
